// MixLinear_GEMM_49177375539613
// MI455X (gfx1250) — 16-bit and 8-bit matrix operand order repaired and checked by reading, compiled, not run on hardware
//
#include <hip/hip_runtime.h>
#include <stdint.h>

typedef _Float16 half_t;
typedef __attribute__((ext_vector_type(16))) _Float16 v16h;
typedef __attribute__((ext_vector_type(8)))  _Float16 v8h;
typedef __attribute__((ext_vector_type(8)))  float    v8f;
typedef __attribute__((ext_vector_type(8)))  int      v8i;
typedef __attribute__((ext_vector_type(4)))  int      v4i;
typedef __attribute__((ext_vector_type(2)))  int      v2i;

#define M_TOTAL 8192
#define K_TOTAL 4096
#define N_TOTAL 4096
#define NOUT    32

#define BM 128
#define BN 128
#define BK 64
#define LDSTR 80

#if __has_builtin(__builtin_amdgcn_global_load_async_to_lds_b128) && \
    __has_builtin(__builtin_amdgcn_s_wait_asynccnt)
#define USE_ASYNC_LDS 1
#else
#define USE_ASYNC_LDS 0
#endif

#if USE_ASYNC_LDS
#define ASYNC_CP_B128(gsrc, ldst) \
    __builtin_amdgcn_global_load_async_to_lds_b128((v4i*)(uintptr_t)(gsrc), (v4i*)(ldst), 0, 0)
#endif

__global__ __launch_bounds__(256) void quant_rows(
    const half_t* __restrict__ x,
    const int*    __restrict__ ind,
    int8_t*       __restrict__ qx,
    float*        __restrict__ srow,
    half_t*       __restrict__ xout)
{
    __shared__ float red[256];
    const int m = blockIdx.x;
    const int t = threadIdx.x;
    const half_t* xr = x + (size_t)m * K_TOTAL;
    const int base = t * 16;

    v8h a = *(const v8h*)(xr + base);
    v8h b = *(const v8h*)(xr + base + 8);
    float vals[16];
#pragma unroll
    for (int i = 0; i < 8; ++i) { vals[i] = (float)a[i]; vals[8 + i] = (float)b[i]; }

    float mx = 0.f;
#pragma unroll
    for (int i = 0; i < 16; ++i) mx = fmaxf(mx, fabsf(vals[i]));
    red[t] = mx;
    __syncthreads();
#pragma unroll
    for (int s = 128; s > 0; s >>= 1) {
        if (t < s) red[t] = fmaxf(red[t], red[t + s]);
        __syncthreads();
    }
    const float amax = red[0];
    const float scale = amax * (1.f / 127.f);
    const float inv   = amax > 0.f ? 127.f / amax : 0.f;
    if (t == 0) srow[m] = scale;

    union { int8_t c[16]; v4i v; } u;
#pragma unroll
    for (int i = 0; i < 16; ++i) {
        float v = vals[i] * inv;
        v = fminf(fmaxf(v, -128.f), 127.f);
        u.c[i] = (int8_t)(int)rintf(v);
    }
    *(v4i*)(qx + (size_t)m * K_TOTAL + base) = u.v;

    if (t < NOUT) xout[(size_t)m * NOUT + t] = xr[ind[t]];
}

__global__ __launch_bounds__(256) void mix_gemm(
    const int8_t* __restrict__ qw,
    const half_t* __restrict__ scol,
    const half_t* __restrict__ wc,
    const int8_t* __restrict__ qx,
    const float*  __restrict__ srow,
    const half_t* __restrict__ xout,
    half_t*       __restrict__ out)
{
    __shared__ __attribute__((aligned(16))) int8_t As[2][BM * LDSTR];
    __shared__ __attribute__((aligned(16))) int8_t Bs[2][BM * LDSTR];
    __shared__ float sR[BM];

    const int t    = threadIdx.x;
    const int lane = t & 31;
    const int wave = t >> 5;
    const int lh   = lane >> 4;
    const int l16  = lane & 15;
    const int wm   = wave >> 2;
    const int wn   = wave & 3;
    const int bm   = blockIdx.y * BM;
    const int bn   = blockIdx.x * BN;

    if (t < BM) sR[t] = srow[bm + t];

    const int lrow = t >> 2;
    const int lcol = (t & 3) * 16;
    const int8_t* gA = qx + (size_t)(bm + lrow) * K_TOTAL + lcol;
    const int8_t* gB = qw + (size_t)(bn + lrow) * K_TOTAL + lcol;
    const int sOff = lrow * LDSTR + lcol;

    v8i acc[4][2];
#pragma unroll
    for (int i = 0; i < 4; ++i)
#pragma unroll
        for (int j = 0; j < 2; ++j)
#pragma unroll
            for (int r = 0; r < 8; ++r) acc[i][j][r] = 0;

    const int NK = K_TOTAL / BK;

#if USE_ASYNC_LDS
    ASYNC_CP_B128(gA,                        &As[0][sOff]);
    ASYNC_CP_B128(gA + (size_t)64 * K_TOTAL, &As[0][sOff + 64 * LDSTR]);
    ASYNC_CP_B128(gB,                        &Bs[0][sOff]);
    ASYNC_CP_B128(gB + (size_t)64 * K_TOTAL, &Bs[0][sOff + 64 * LDSTR]);
#else
    v4i ra0 = *(const v4i*)(gA);
    v4i ra1 = *(const v4i*)(gA + (size_t)64 * K_TOTAL);
    v4i rb0 = *(const v4i*)(gB);
    v4i rb1 = *(const v4i*)(gB + (size_t)64 * K_TOTAL);
    *(v4i*)&As[0][sOff]              = ra0;
    *(v4i*)&As[0][sOff + 64 * LDSTR] = ra1;
    *(v4i*)&Bs[0][sOff]              = rb0;
    *(v4i*)&Bs[0][sOff + 64 * LDSTR] = rb1;
    __syncthreads();
#endif

    for (int kt = 0; kt < NK; ++kt) {
        const int buf = kt & 1;

#if USE_ASYNC_LDS
        if (kt + 1 < NK) {
            const int nb = buf ^ 1;
            const int8_t* pA = gA + (size_t)(kt + 1) * BK;
            const int8_t* pB = gB + (size_t)(kt + 1) * BK;
            ASYNC_CP_B128(pA,                        &As[nb][sOff]);
            ASYNC_CP_B128(pA + (size_t)64 * K_TOTAL, &As[nb][sOff + 64 * LDSTR]);
            ASYNC_CP_B128(pB,                        &Bs[nb][sOff]);
            ASYNC_CP_B128(pB + (size_t)64 * K_TOTAL, &Bs[nb][sOff + 64 * LDSTR]);
            __builtin_amdgcn_s_wait_asynccnt(4);
        } else {
            __builtin_amdgcn_s_wait_asynccnt(0);
        }
        __syncthreads();
#else
        if (kt + 1 < NK) {
            const int8_t* pA = gA + (size_t)(kt + 1) * BK;
            const int8_t* pB = gB + (size_t)(kt + 1) * BK;
            ra0 = *(const v4i*)(pA);
            ra1 = *(const v4i*)(pA + (size_t)64 * K_TOTAL);
            rb0 = *(const v4i*)(pB);
            rb1 = *(const v4i*)(pB + (size_t)64 * K_TOTAL);
        }
#endif
        if (kt + 2 < NK) {
            __builtin_prefetch(gA + (size_t)(kt + 2) * BK, 0, 1);
            __builtin_prefetch(gB + (size_t)(kt + 2) * BK, 0, 1);
        }

        v8i bfrag[2];
#pragma unroll
        for (int nt = 0; nt < 2; ++nt) {
            const int nrow = wn * 32 + nt * 16 + l16;
            const int bb   = nrow * LDSTR + lh * 16;
            v4i lo = *(const v4i*)&Bs[buf][bb];
            v4i hi = *(const v4i*)&Bs[buf][bb + 32];
#pragma unroll
            for (int i = 0; i < 4; ++i) { bfrag[nt][i] = lo[i]; bfrag[nt][4 + i] = hi[i]; }
        }

#pragma unroll
        for (int mt = 0; mt < 4; ++mt) {
            const int mrow = wm * 64 + mt * 16 + l16;
            v8i afrag;
#pragma unroll
            for (int p = 0; p < 4; ++p) {
                v2i d = *(const v2i*)&As[buf][mrow * LDSTR + (p >> 1) * 32 + lh * 16 + (p & 1) * 8];
                afrag[2 * p]     = d[0];
                afrag[2 * p + 1] = d[1];
            }
#pragma unroll
            for (int nt = 0; nt < 2; ++nt) {
                acc[mt][nt] = __builtin_amdgcn_wmma_i32_16x16x64_iu8(
                     true, afrag,  true, bfrag[nt],
                    acc[mt][nt],  false,  false);
            }
        }

#if USE_ASYNC_LDS
        __syncthreads();
#else
        if (kt + 1 < NK) {
            const int nb = buf ^ 1;
            *(v4i*)&As[nb][sOff]              = ra0;
            *(v4i*)&As[nb][sOff + 64 * LDSTR] = ra1;
            *(v4i*)&Bs[nb][sOff]              = rb0;
            *(v4i*)&Bs[nb][sOff + 64 * LDSTR] = rb1;
        }
        __syncthreads();
#endif
    }

    v16h aout[4];
#pragma unroll
    for (int mt = 0; mt < 4; ++mt) {
        const int mg = bm + wm * 64 + mt * 16 + l16;
        const half_t* xr = xout + (size_t)mg * NOUT;
        v8h lo = *(const v8h*)(xr + lh * 8);
        v8h hi = *(const v8h*)(xr + 16 + lh * 8);
#pragma unroll
        for (int i = 0; i < 8; ++i) { aout[mt][i] = lo[i]; aout[mt][8 + i] = hi[i]; }
    }
    v16h bout[2];
    float wsc[2], wbi[2];
#pragma unroll
    for (int nt = 0; nt < 2; ++nt) {
        const int ng = bn + wn * 32 + nt * 16 + l16;
        v8h blo = *(const v8h*)(wc + (size_t)ng * NOUT + lh * 8);
        v8h bhi = *(const v8h*)(wc + (size_t)ng * NOUT + 16 + lh * 8);
#pragma unroll
        for (int i = 0; i < 8; ++i) { bout[nt][i] = blo[i]; bout[nt][8 + i] = bhi[i]; }
        wsc[nt] = (float)scol[2 * ng];
        wbi[nt] = (float)scol[2 * ng + 1];
    }

#pragma unroll
    for (int mt = 0; mt < 4; ++mt) {
#pragma unroll
        for (int nt = 0; nt < 2; ++nt) {
            v8f corr = {0.f, 0.f, 0.f, 0.f, 0.f, 0.f, 0.f, 0.f};
            corr = __builtin_amdgcn_wmma_f32_16x16x32_f16(
                false, aout[mt], false, bout[nt], (short)0, corr, false, false);
#pragma unroll
            for (int r = 0; r < 8; ++r) {
                const int mloc = wm * 64 + mt * 16 + r + lh * 8;
                const int mg   = bm + mloc;
                const int ng   = bn + wn * 32 + nt * 16 + l16;
                float y = (float)acc[mt][nt][r] * sR[mloc] * wsc[nt] + wbi[nt] + corr[r];
                out[(size_t)mg * N_TOTAL + ng] = (half_t)y;
            }
        }
    }
}

extern "C" void kernel_launch(void* const* d_in, const int* in_sizes, int n_in,
                              void* d_out, int out_size, void* d_ws, size_t ws_size,
                              hipStream_t stream) {
    (void)in_sizes; (void)n_in; (void)out_size; (void)ws_size;
    const half_t* x    = (const half_t*)d_in[0];
    const int8_t* qw   = (const int8_t*)d_in[1];
    const half_t* scol = (const half_t*)d_in[2];
    const half_t* wc   = (const half_t*)d_in[3];
    const int*    ind  = (const int*)d_in[4];
    half_t* out = (half_t*)d_out;

    int8_t* qx   = (int8_t*)d_ws;
    float*  srow = (float*)((char*)d_ws + (size_t)M_TOTAL * K_TOTAL);
    half_t* xo   = (half_t*)((char*)d_ws + (size_t)M_TOTAL * K_TOTAL + (size_t)M_TOTAL * 4);

    quant_rows<<<M_TOTAL, 256, 0, stream>>>(x, ind, qx, srow, xo);

    dim3 grid(N_TOTAL / BN, M_TOTAL / BM);
    mix_gemm<<<grid, 256, 0, stream>>>(qw, scol, wc, qx, srow, xo, out);
}
